// MultiHeadAttention_66838281060488
// MI455X (gfx1250) — matrix operand order repaired and checked by reading, not compiled, not run on hardware
//
#include <hip/hip_runtime.h>

typedef __bf16 bf16;
typedef bf16  v16bf __attribute__((ext_vector_type(16)));
typedef bf16  v8bf  __attribute__((ext_vector_type(8)));
typedef float v8f   __attribute__((ext_vector_type(8)));

static constexpr int kH = 12, kD = 768, kDH = 64, kB = 2, kS = 2048;
static constexpr int kLDW = kD + 8;  // padded LDS row stride (388 dwords -> conflict-free)

__device__ __forceinline__ v8f wmma_bf16(v16bf a, v16bf b, v8f c) {
  // D = A(16x32 bf16) * B(32x16 bf16) + C(16x16 f32)
  return __builtin_amdgcn_wmma_f32_16x16x32_bf16(false, a, false, b, (short)0, c,
                                                 false, false);
}

// A fragment: 16x32 tile from row-major [.., ld] bf16; rows row0..+15, K=k0..+31.
// ISA 7.12.2: lane holds row lane%16; K chunks g*8..+7 and 16+g*8..+7.
__device__ __forceinline__ v16bf load_afrag(const bf16* __restrict__ A, int ld,
                                            int row0, int k0, int lane) {
  int r = row0 + (lane & 15);
  int g = lane >> 4;
  const bf16* p = A + (size_t)r * ld + k0 + g * 8;
  v8bf lo = *(const v8bf*)p;
  v8bf hi = *(const v8bf*)(p + 16);
  v16bf out;
#pragma unroll
  for (int i = 0; i < 8; ++i) { out[i] = lo[i]; out[i + 8] = hi[i]; }
  return out;
}

// B fragment from global: B is 32(K) x 16(N); stored K-contiguous as BT[N][K].
// ISA: lane holds col lane%16; K chunks g*8..+7 and 16+g*8..+7.
__device__ __forceinline__ v16bf load_bfrag(const bf16* __restrict__ BT, int ld,
                                            int n0, int k0, int lane) {
  int n = n0 + (lane & 15);
  int g = lane >> 4;
  const bf16* p = BT + (size_t)n * ld + k0 + g * 8;
  v8bf lo = *(const v8bf*)p;
  v8bf hi = *(const v8bf*)(p + 16);
  v16bf out;
#pragma unroll
  for (int i = 0; i < 8; ++i) { out[i] = lo[i]; out[i + 8] = hi[i]; }
  return out;
}

// B fragment from an LDS-staged BT tile with padded row stride kLDW (two ds_b128).
__device__ __forceinline__ v16bf load_bfrag_lds(const bf16* BT, int n0, int k0, int lane) {
  int n = n0 + (lane & 15);
  int g = lane >> 4;
  const bf16* p = BT + n * kLDW + k0 + g * 8;
  v8bf lo = *(const v8bf*)p;
  v8bf hi = *(const v8bf*)(p + 16);
  v16bf out;
#pragma unroll
  for (int i = 0; i < 8; ++i) { out[i] = lo[i]; out[i + 8] = hi[i]; }
  return out;
}

// ---------------- precompute: f32 -> bf16 (optionally transposed) ----------------
__global__ void k_f32_to_bf16(const float* __restrict__ src, bf16* __restrict__ dst, int n) {
  for (int i = blockIdx.x * blockDim.x + threadIdx.x; i < n; i += gridDim.x * blockDim.x)
    dst[i] = (bf16)src[i];
}

// src: [nh, R, C] f32  ->  dst: [nh, C, R] bf16
__global__ void k_transpose_bf16(const float* __restrict__ src, bf16* __restrict__ dst,
                                 int nh, int R, int C) {
  int total = nh * R * C;
  for (int i = blockIdx.x * blockDim.x + threadIdx.x; i < total; i += gridDim.x * blockDim.x) {
    int h = i / (R * C);
    int rem = i - h * R * C;
    int r = rem / C, c = rem - r * C;
    dst[(size_t)h * R * C + (size_t)c * R + r] = (bf16)src[i];
  }
}

// Cooperatively stage a [64, 768] K-contiguous BT tile into padded LDS.
__device__ __forceinline__ void stage_bt(bf16* smem, const bf16* __restrict__ BT, int tid) {
  constexpr int CHUNKS = kD / 8;  // 96 x v8bf per row
  for (int c = tid; c < kDH * CHUNKS; c += 256) {
    int row = c / CHUNKS;
    int kc = c - row * CHUNKS;
    *(v8bf*)(smem + row * kLDW + kc * 8) = *(const v8bf*)(BT + (size_t)row * kD + kc * 8);
  }
  __syncthreads();
}

// ---------------- QKV projection: [S,768] x [768,64] per (b,h) -------------------
// mode 0: Q, row-major [B,H,S,64], scaled by 1/sqrt(64)
// mode 1: K, row-major [B,H,S,64]
// mode 2: V, transposed [B,H,64,S]
__global__ __launch_bounds__(256) void k_qkv(const bf16* __restrict__ xbf,
                                             const bf16* __restrict__ wt,
                                             const float* __restrict__ bias,
                                             bf16* __restrict__ out, int mode) {
  extern __shared__ bf16 smem[];  // [64][kLDW]
  int tid = threadIdx.x, lane = tid & 31, w = tid >> 5;
  int g = lane >> 4, col = lane & 15;
  int h = blockIdx.y, b = blockIdx.z;
  int m0 = blockIdx.x * 128 + w * 16;
  const bf16* A = xbf + (size_t)b * kS * kD;       // [S, 768] row-major
  stage_bt(smem, wt + (size_t)h * kDH * kD, tid);  // W^T head -> LDS

  v8f acc[4] = {};
  // software pipeline: fragments for step k are loaded during step k-32
  v16bf a_cur = load_afrag(A, kD, m0, 0, lane);
  v16bf b_cur[4];
#pragma unroll
  for (int n = 0; n < 4; ++n) b_cur[n] = load_bfrag_lds(smem, n * 16, 0, lane);

  for (int k0 = 0; k0 < kD; k0 += 32) {
    int kn = (k0 + 32 < kD) ? (k0 + 32) : 0;        // wrap: loads stay unconditional
    v16bf a_nxt = load_afrag(A, kD, m0, kn, lane);  // next-A (global)
    v16bf b_nxt[4];                                  // next-B (LDS)
#pragma unroll
    for (int n = 0; n < 4; ++n) b_nxt[n] = load_bfrag_lds(smem, n * 16, kn, lane);
#pragma unroll
    for (int n = 0; n < 4; ++n) acc[n] = wmma_bf16(a_cur, b_cur[n], acc[n]);
    a_cur = a_nxt;
#pragma unroll
    for (int n = 0; n < 4; ++n) b_cur[n] = b_nxt[n];
  }

  size_t bh = (size_t)b * kH + h;
  const float* bs = bias + h * kDH;
  if (mode == 2) {
    // C layout: lane has col e fixed and 8 consecutive rows t = m0+8g+r -> 16B store
#pragma unroll
    for (int n = 0; n < 4; ++n) {
      int e = n * 16 + col;
      float bb = bs[e];
      v8bf pk;
#pragma unroll
      for (int r = 0; r < 8; ++r) pk[r] = (bf16)(acc[n][r] + bb);
      *(v8bf*)(out + (bh * kDH + e) * kS + m0 + g * 8) = pk;
    }
  } else {
    float sc = (mode == 0) ? 0.125f : 1.0f;  // fold 1/sqrt(DH) into Q
#pragma unroll
    for (int n = 0; n < 4; ++n) {
      int e = n * 16 + col;
      float bb = bs[e];
#pragma unroll
      for (int r = 0; r < 8; ++r)
        out[(bh * kS + m0 + 8 * g + r) * kDH + e] = (bf16)((acc[n][r] + bb) * sc);
    }
  }
}

// ---------------- Flash attention: per (b,h), 16 query rows per wave -------------
__global__ __launch_bounds__(256) void k_attn(const bf16* __restrict__ Q,
                                              const bf16* __restrict__ K,
                                              const bf16* __restrict__ VT,
                                              bf16* __restrict__ ctx) {
  __shared__ bf16 ldsP[8][16][48];  // per-wave 16x32 P tile, row stride 48 (96B)
  int tid = threadIdx.x, lane = tid & 31, w = tid >> 5;
  int g = lane >> 4, col = lane & 15;
  int h = blockIdx.y, b = blockIdx.z;
  int m0 = blockIdx.x * 128 + w * 16;
  size_t bh = (size_t)b * kH + h;
  const bf16* Qp = Q + bh * (size_t)kS * kDH;   // [S,64] row-major
  const bf16* Kp = K + bh * (size_t)kS * kDH;   // [S,64] row-major (== BT for QK^T)
  const bf16* Vp = VT + bh * (size_t)kDH * kS;  // [64,S]          (== BT for P*V)

  v16bf q0 = load_afrag(Qp, kDH, m0, 0, lane);
  v16bf q1 = load_afrag(Qp, kDH, m0, 32, lane);

  float rmax[8], rsum[8];
  v8f acc[4] = {};
#pragma unroll
  for (int r = 0; r < 8; ++r) { rmax[r] = -3.0e38f; rsum[r] = 0.f; }

  // pipelined K fragments for j = 0
  v16bf kf[4];
  kf[0] = load_bfrag(Kp, kDH, 0, 0, lane);
  kf[1] = load_bfrag(Kp, kDH, 0, 32, lane);
  kf[2] = load_bfrag(Kp, kDH, 16, 0, lane);
  kf[3] = load_bfrag(Kp, kDH, 16, 32, lane);

  for (int j = 0; j < kS; j += 32) {
    // scores: 16x32 tile = two 16x16 C tiles, K-dim = 64 (2 WMMAs each)
    v8f s0 = {}, s1 = {};
    s0 = wmma_bf16(q0, kf[0], s0);
    s0 = wmma_bf16(q1, kf[1], s0);
    s1 = wmma_bf16(q0, kf[2], s1);
    s1 = wmma_bf16(q1, kf[3], s1);

    // issue this block's V fragments and next block's K fragments NOW,
    // so they overlap the softmax VALU chain + LDS transpose below
    v16bf vf[4];
#pragma unroll
    for (int n = 0; n < 4; ++n) vf[n] = load_bfrag(Vp, kS, n * 16, j, lane);
    int jn = (j + 32 < kS) ? (j + 32) : 0;  // wrap keeps loads branch-free
    kf[0] = load_bfrag(Kp, kDH, jn, 0, lane);
    kf[1] = load_bfrag(Kp, kDH, jn, 32, lane);
    kf[2] = load_bfrag(Kp, kDH, jn + 16, 0, lane);
    kf[3] = load_bfrag(Kp, kDH, jn + 16, 32, lane);
    if (j + 64 < kS) __builtin_prefetch(Kp + (size_t)(j + 64) * kDH, 0, 0);

    // online softmax; row r+8g lives across the 16 lanes of this half-wave
#pragma unroll
    for (int r = 0; r < 8; ++r) {
      float mx = fmaxf(s0[r], s1[r]);
#pragma unroll
      for (int o = 1; o < 16; o <<= 1) mx = fmaxf(mx, __shfl_xor(mx, o, 32));
      float mnew = fmaxf(rmax[r], mx);
      float alpha = __expf(rmax[r] - mnew);
      float p0 = __expf(s0[r] - mnew);
      float p1 = __expf(s1[r] - mnew);
      float rs = p0 + p1;
#pragma unroll
      for (int o = 1; o < 16; o <<= 1) rs += __shfl_xor(rs, o, 32);
      rsum[r] = rsum[r] * alpha + rs;
      rmax[r] = mnew;
      s0[r] = p0; s1[r] = p1;
#pragma unroll
      for (int n = 0; n < 4; ++n) acc[n][r] = acc[n][r] * alpha;
    }

    // C-layout -> A-layout transpose of P through LDS (per-wave region)
#pragma unroll
    for (int r = 0; r < 8; ++r) {
      ldsP[w][r + 8 * g][col]      = (bf16)s0[r];
      ldsP[w][r + 8 * g][16 + col] = (bf16)s1[r];
    }
    asm volatile("s_wait_dscnt 0x0" ::: "memory");
    __builtin_amdgcn_wave_barrier();
    v16bf pf;
    {
      int row = lane & 15;
      v8bf lo = *(const v8bf*)&ldsP[w][row][g * 8];
      v8bf hi = *(const v8bf*)&ldsP[w][row][16 + g * 8];
#pragma unroll
      for (int i = 0; i < 8; ++i) { pf[i] = lo[i]; pf[i + 8] = hi[i]; }
    }
    __builtin_amdgcn_wave_barrier();

    // ctx += P(16x32) * V(32x64)
#pragma unroll
    for (int n = 0; n < 4; ++n) acc[n] = wmma_bf16(pf, vf[n], acc[n]);
  }

  // epilogue: normalize, write ctx as [B,S,H*64] (heads already concatenated)
#pragma unroll
  for (int r = 0; r < 8; ++r) rsum[r] = 1.0f / rsum[r];
#pragma unroll
  for (int n = 0; n < 4; ++n) {
    int e = h * kDH + n * 16 + col;
#pragma unroll
    for (int r = 0; r < 8; ++r) {
      int s = m0 + 8 * g + r;
      ctx[((size_t)b * kS + s) * kD + e] = (bf16)(acc[n][r] * rsum[r]);
    }
  }
}

// ---------------- Output projection: [4096,768] x [768,768] + bo -----------------
__global__ __launch_bounds__(256) void k_proj(const bf16* __restrict__ ctx,
                                              const bf16* __restrict__ WoT,
                                              const float* __restrict__ bo,
                                              float* __restrict__ out) {
  extern __shared__ bf16 smem[];  // [64][kLDW]
  int tid = threadIdx.x, lane = tid & 31, w = tid >> 5;
  int g = lane >> 4, col = lane & 15;
  int m0 = blockIdx.x * 128 + w * 16;
  int n0 = blockIdx.y * 64;
  stage_bt(smem, WoT + (size_t)n0 * kD, tid);  // 64 output cols of Wo^T -> LDS

  v8f acc[4] = {};
  v16bf a_cur = load_afrag(ctx, kD, m0, 0, lane);
  v16bf b_cur[4];
#pragma unroll
  for (int n = 0; n < 4; ++n) b_cur[n] = load_bfrag_lds(smem, n * 16, 0, lane);

  for (int k0 = 0; k0 < kD; k0 += 32) {
    int kn = (k0 + 32 < kD) ? (k0 + 32) : 0;
    v16bf a_nxt = load_afrag(ctx, kD, m0, kn, lane);
    v16bf b_nxt[4];
#pragma unroll
    for (int n = 0; n < 4; ++n) b_nxt[n] = load_bfrag_lds(smem, n * 16, kn, lane);
#pragma unroll
    for (int n = 0; n < 4; ++n) acc[n] = wmma_bf16(a_cur, b_cur[n], acc[n]);
    a_cur = a_nxt;
#pragma unroll
    for (int n = 0; n < 4; ++n) b_cur[n] = b_nxt[n];
  }
#pragma unroll
  for (int n = 0; n < 4; ++n) {
    int e = n0 + n * 16 + col;
    float bb = bo[e];
#pragma unroll
    for (int r = 0; r < 8; ++r)
      out[(size_t)(m0 + 8 * g + r) * kD + e] = acc[n][r] + bb;
  }
}

extern "C" void kernel_launch(void* const* d_in, const int* in_sizes, int n_in,
                              void* d_out, int out_size, void* d_ws, size_t ws_size,
                              hipStream_t stream) {
  (void)in_sizes; (void)n_in; (void)out_size; (void)ws_size;
  const float* x  = (const float*)d_in[0];
  const float* Wq = (const float*)d_in[1];
  const float* Wk = (const float*)d_in[2];
  const float* Wv = (const float*)d_in[3];
  const float* bq = (const float*)d_in[4];
  const float* bk = (const float*)d_in[5];
  const float* bv = (const float*)d_in[6];
  const float* Wo = (const float*)d_in[7];
  const float* bo = (const float*)d_in[8];
  float* out = (float*)d_out;

  char* ws = (char*)d_ws;
  size_t off = 0;
  auto alloc = [&](size_t bytes) -> void* {
    void* p = ws + off;
    off = (off + bytes + 255) & ~(size_t)255;
    return p;
  };
  bf16* xbf  = (bf16*)alloc((size_t)kB * kS * kD * 2);        // x in bf16
  bf16* wqt  = (bf16*)alloc((size_t)kH * kD * kDH * 2);       // Wq^T [H,64,768]
  bf16* wkt  = (bf16*)alloc((size_t)kH * kD * kDH * 2);
  bf16* wvt  = (bf16*)alloc((size_t)kH * kD * kDH * 2);
  bf16* wot  = (bf16*)alloc((size_t)kD * kD * 2);             // Wo^T [768,768]
  bf16* qb   = (bf16*)alloc((size_t)kB * kH * kS * kDH * 2);  // [B,H,S,64]
  bf16* kb_  = (bf16*)alloc((size_t)kB * kH * kS * kDH * 2);  // [B,H,S,64]
  bf16* vtb  = (bf16*)alloc((size_t)kB * kH * kDH * kS * 2);  // [B,H,64,S]
  bf16* ctxb = (bf16*)alloc((size_t)kB * kS * kD * 2);        // [B,S,768]

  k_f32_to_bf16<<<512, 256, 0, stream>>>(x, xbf, kB * kS * kD);
  k_transpose_bf16<<<512, 256, 0, stream>>>(Wq, wqt, kH, kD, kDH);
  k_transpose_bf16<<<512, 256, 0, stream>>>(Wk, wkt, kH, kD, kDH);
  k_transpose_bf16<<<512, 256, 0, stream>>>(Wv, wvt, kH, kD, kDH);
  k_transpose_bf16<<<512, 256, 0, stream>>>(Wo, wot, 1, kD, kD);

  size_t wlds = (size_t)kDH * kLDW * 2;  // 99,328 B staged weight tile
  dim3 gq(kS / 128, kH, kB);
  k_qkv<<<gq, 256, wlds, stream>>>(xbf, wqt, bq, qb, 0);
  k_qkv<<<gq, 256, wlds, stream>>>(xbf, wkt, bk, kb_, 1);
  k_qkv<<<gq, 256, wlds, stream>>>(xbf, wvt, bv, vtb, 2);

  k_attn<<<dim3(kS / 128, kH, kB), 256, 0, stream>>>(qb, kb_, vtb, ctxb);

  k_proj<<<dim3((kB * kS) / 128, kD / 64, 1), 256, wlds, stream>>>(ctxb, wot, bo, out);
}
